// DownResBlock_5549097746511
// MI455X (gfx1250) — matrix operand order repaired and checked by reading, not compiled, not run on hardware
//
#include <hip/hip_runtime.h>

// ---------------------------------------------------------------------------
// MI455X / gfx1250 implementation of the masked DownResBlock.
// All convs are implicit GEMMs on v_wmma_f32_16x16x32_bf16 (bf16 A/B, f32 acc).
// Activations live channel-last in workspace; weights are pre-shuffled into
// a contiguous-k A-fragment order (k = 16*half + e, as B uses) so A fragments are contiguous loads.
// ---------------------------------------------------------------------------

typedef __attribute__((ext_vector_type(16))) __bf16 v16bf;
typedef __attribute__((ext_vector_type(8)))  float  v8f;

constexpr int DD = 160, HH = 160, WW = 32;
constexpr int NVOXEL = DD * HH * WW;       // 819200
constexpr int COUT = 64;

// --------------------------- stats reset -----------------------------------
// stats layout (floats): [0] = n_active, [64..127] = S1, [128..191] = S2
__global__ void k_reset(float* stats, int zero_all) {
  int t = threadIdx.x;                     // 192 threads
  if (zero_all || t >= 64) stats[t] = 0.f;
}

// ------------------- weight shuffle: fp32 OIDHW -> bf16 WMMA-A --------------
// dst index = (((ctile*TAPS + tap)*KBLK + kb)*32 + lane)*16 + e
// lane: m = lane&15 (row M), half = lane>>4 selects K range 16*half..16*half+15 (contiguous k).
__global__ void k_prep_w(const float* __restrict__ src, __bf16* __restrict__ dst,
                         int CIN, int TAPS) {
  int idx = blockIdx.x * 128 + threadIdx.x;
  int KBLK = CIN / 32;
  int total = 4 * TAPS * KBLK * 32;
  if (idx >= total) return;
  int lane = idx & 31;
  int t = idx >> 5;
  int kb = t % KBLK;  t /= KBLK;
  int tap = t % TAPS; t /= TAPS;
  int ctile = t;
  int m = lane & 15, half = lane >> 4;
  int cout = ctile * 16 + m;
  for (int e = 0; e < 16; ++e) {
    int j = e >> 1, b = e & 1;
    int kl = 16 * half + 2 * j + b;
    int cin = kb * 32 + kl;
    float w = src[((size_t)cout * CIN + cin) * TAPS + tap];
    dst[(size_t)idx * 16 + e] = (__bf16)w;
  }
}

// -------------- mask count + x*m -> bf16 channel-last convert ---------------
__global__ void k_mask_convert(const float* __restrict__ x,
                               const unsigned char* __restrict__ mask,
                               __bf16* __restrict__ xs_bf,
                               float* __restrict__ stats) {
  int v = blockIdx.x * 256 + threadIdx.x;
  __shared__ float cnt;
  if (threadIdx.x == 0) cnt = 0.f;
  __syncthreads();
  float m = mask[v] ? 1.f : 0.f;
  for (int c = 0; c < 32; ++c)
    xs_bf[(size_t)v * 32 + c] = (__bf16)(x[(size_t)c * NVOXEL + v] * m);
  atomicAdd(&cnt, m);                       // ds_add_f32
  __syncthreads();
  if (threadIdx.x == 0) atomicAdd(&stats[0], cnt);
}

// ------------------------------ conv kernel ---------------------------------
// EPI==0 : stride-1 conv -> lrelu -> *mask -> h32 (channel-last) + S1/S2 stats
// EPI==1 : stride (2,2,1) pool conv -> *out_mask -> c-major NCDHW output
template <int CIN, int KD, int KH, int SDH, int EPI>
__global__ __launch_bounds__(128) void k_conv(
    const __bf16* __restrict__ in,          // [D][H][W][CIN] bf16
    const __bf16* __restrict__ wt,          // WMMA-A shuffled weights
    const unsigned char* __restrict__ mask, // [D][H][W]
    float* __restrict__ h32,                // EPI==0 out
    float* __restrict__ stats,              // EPI==0 S1/S2
    float* __restrict__ out_cm)             // EPI==1 out (c-major)
{
  constexpr int KW = 3, TAPS = KD * KH * KW, KBLK = CIN / 32, PW = 18;
  constexpr int NPVOX = KD * KH * PW;
  __shared__ __bf16 patch[NPVOX * CIN];
  __shared__ float bS[128];

  const int tid = threadIdx.x;
  const int lane = tid & 31, wave = tid >> 5;

  int t = blockIdx.x;
  const int wbase = (t & 1) * 16; t >>= 1;
  constexpr int HOUT = (SDH == 1) ? HH : HH / 2;
  const int ho  = t % HOUT;
  const int d_o = t / HOUT;
  const int d0 = (SDH == 1) ? (d_o - KD / 2) : (2 * d_o - 1);
  const int h0 = (SDH == 1) ? (ho  - KH / 2) : (2 * ho  - 1);
  const int w0 = wbase - 1;

  if constexpr (EPI == 0) bS[tid] = 0.f;

  __builtin_prefetch(wt + (size_t)wave * TAPS * KBLK * 32 * 16, 0, 1);

  // cooperative halo-patch load: zero-padded bf16, 16B chunks, channel-last
  constexpr int CPV = CIN / 8;               // uint4 chunks per patch voxel
  for (int ch = tid; ch < NPVOX * CPV; ch += 128) {
    int vox = ch / CPV, cc = ch % CPV;
    int pw = vox % PW; int r = vox / PW;
    int ph = r % KH;   int pd = r / KH;
    int gd = d0 + pd, gh = h0 + ph, gw = w0 + pw;
    uint4 val = make_uint4(0u, 0u, 0u, 0u);
    if (gd >= 0 && gd < DD && gh >= 0 && gh < HH && gw >= 0 && gw < WW)
      val = ((const uint4*)(in + (size_t)((gd * HH + gh) * WW + gw) * CIN))[cc];
    ((uint4*)(patch + (size_t)vox * CIN))[cc] = val;
  }
  __syncthreads();

  const int n = lane & 15, half = lane >> 4;
  v8f acc = {};
#pragma unroll
  for (int tap = 0; tap < TAPS; ++tap) {
    const int kd = tap / (KH * KW);
    const int kh = (tap / KW) % KH;
    const int kw = tap % KW;
    const int pvox = (kd * KH + kh) * PW + n + kw;
#pragma unroll
    for (int kb = 0; kb < KBLK; ++kb) {
      v16bf bm = *(const v16bf*)(patch + (size_t)pvox * CIN + kb * 32 + half * 16);
      v16bf am = *(const v16bf*)(wt +
          ((((size_t)wave * TAPS + tap) * KBLK + kb) * 32 + lane) * 16);
      acc = __builtin_amdgcn_wmma_f32_16x16x32_bf16(
          false, am, false, bm, (short)0, acc, false, false);
    }
  }

  const int c0 = wave * 16 + half * 8;       // 8 contiguous couts per lane
  if constexpr (EPI == 0) {
    const int v = (d_o * HH + ho) * WW + wbase + n;
    const float m = mask[v] ? 1.f : 0.f;
    float hv[8];
#pragma unroll
    for (int r = 0; r < 8; ++r) {
      float xv = acc[r];
      xv = (xv >= 0.f) ? xv : 0.01f * xv;    // leaky relu
      hv[r] = xv * m;
    }
    float4* dst = (float4*)(h32 + (size_t)v * COUT + c0);
    dst[0] = make_float4(hv[0], hv[1], hv[2], hv[3]);
    dst[1] = make_float4(hv[4], hv[5], hv[6], hv[7]);
#pragma unroll
    for (int r = 0; r < 8; ++r) {
      atomicAdd(&bS[c0 + r], hv[r]);
      atomicAdd(&bS[64 + c0 + r], hv[r] * hv[r]);
    }
    __syncthreads();
    if (tid < 64) {
      atomicAdd(&stats[64 + tid], bS[tid]);
      atomicAdd(&stats[128 + tid], bS[64 + tid]);
    }
  } else {
    constexpr int DOUT = DD / 2;
    const int wo = wbase + n;
    bool any = false;
    for (int kd = 0; kd < 3; ++kd)
      for (int kh = 0; kh < 3; ++kh)
        for (int kw = 0; kw < 3; ++kw) {
          int gd = 2 * d_o - 1 + kd, gh = 2 * ho - 1 + kh, gw = wo - 1 + kw;
          if (gd >= 0 && gd < DD && gh >= 0 && gh < HH && gw >= 0 && gw < WW)
            any = any || (mask[(gd * HH + gh) * WW + gw] != 0);
        }
    const float om = any ? 1.f : 0.f;
    const size_t sp = (size_t)(d_o * HOUT + ho) * WW + wo;
#pragma unroll
    for (int r = 0; r < 8; ++r)
      out_cm[(size_t)(c0 + r) * ((size_t)DOUT * HOUT * WW) + sp] = acc[r] * om;
  }
}

// ------------------------------ BN apply ------------------------------------
__global__ void k_bn(const float* __restrict__ h32,
                     const unsigned char* __restrict__ mask,
                     const float* __restrict__ stats,
                     const float* __restrict__ gamma,
                     const float* __restrict__ beta,
                     const float* __restrict__ add_cl,  // optional residual (chan-last)
                     __bf16* __restrict__ out_bf,       // optional bf16 chan-last
                     float* __restrict__ out_cl,        // optional f32 chan-last
                     float* __restrict__ out_cm)        // optional f32 c-major
{
  size_t idx = (size_t)blockIdx.x * 256 + threadIdx.x;  // over NVOXEL*64
  int c = (int)(idx & 63);
  size_t v = idx >> 6;
  float nact = stats[0];
  float mu   = stats[64 + c] / nact;
  float var  = stats[128 + c] / nact - mu * mu;
  float inv  = rsqrtf(var + 1e-5f);
  float m = mask[v] ? 1.f : 0.f;
  float o = (h32[idx] - mu) * inv * gamma[c] + beta[c];
  o *= m;
  if (add_cl) o += add_cl[idx];
  if (out_bf) out_bf[idx] = (__bf16)o;
  if (out_cl) out_cl[idx] = o;
  if (out_cm) out_cm[(size_t)c * NVOXEL + v] = o;
}

// ------------------------------- launcher -----------------------------------
extern "C" void kernel_launch(void* const* d_in, const int* in_sizes, int n_in,
                              void* d_out, int out_size, void* d_ws, size_t ws_size,
                              hipStream_t stream) {
  const float* x  = (const float*)d_in[0];
  const unsigned char* mask = (const unsigned char*)d_in[1];
  const float* WA1 = (const float*)d_in[2];
  const float* WA2 = (const float*)d_in[3];
  const float* WB1 = (const float*)d_in[4];
  const float* WB2 = (const float*)d_in[5];
  const float* WP  = (const float*)d_in[6];
  const float* gA1 = (const float*)d_in[7],  *bA1 = (const float*)d_in[8];
  const float* gA2 = (const float*)d_in[9],  *bA2 = (const float*)d_in[10];
  const float* gB1 = (const float*)d_in[11], *bB1 = (const float*)d_in[12];
  const float* gB2 = (const float*)d_in[13], *bB2 = (const float*)d_in[14];

  char* w = (char*)d_ws;
  float* stats = (float*)w;
  size_t off = 1024;
  __bf16* wA1 = (__bf16*)(w + off); off += (size_t)18432 * 2;   // 4*9*1*32*16
  __bf16* wA2 = (__bf16*)(w + off); off += (size_t)36864 * 2;   // 4*9*2*32*16
  __bf16* wB1 = (__bf16*)(w + off); off += (size_t)18432 * 2;
  __bf16* wB2 = (__bf16*)(w + off); off += (size_t)36864 * 2;
  __bf16* wP  = (__bf16*)(w + off); off += (size_t)110592 * 2;  // 4*27*2*32*16
  off = (off + 255) & ~(size_t)255;
  __bf16* xs   = (__bf16*)(w + off); off += (size_t)NVOXEL * 32 * 2;
  __bf16* act  = (__bf16*)(w + off); off += (size_t)NVOXEL * 64 * 2;
  float*  hbuf = (float*) (w + off); off += (size_t)NVOXEL * 64 * 4;
  float*  resA = (float*) (w + off); off += (size_t)NVOXEL * 64 * 4;

  float* down = (float*)d_out;                          // (1,64,80,80,32)
  float* resB = down + (size_t)64 * 80 * 80 * 32;       // (1,64,160,160,32)

  const int NTILE  = DD * HH * 2;        // 51200 stride-1 tiles
  const int NTILEP = 80 * 80 * 2;        // 12800 pool tiles
  const int NBN    = (int)((size_t)NVOXEL * 64 / 256);  // 204800

  k_reset<<<1, 192, 0, stream>>>(stats, 1);
  k_prep_w<<<(1152 + 127) / 128, 128, 0, stream>>>(WA1, wA1, 32, 9);
  k_prep_w<<<(2304 + 127) / 128, 128, 0, stream>>>(WA2, wA2, 64, 9);
  k_prep_w<<<(1152 + 127) / 128, 128, 0, stream>>>(WB1, wB1, 32, 9);
  k_prep_w<<<(2304 + 127) / 128, 128, 0, stream>>>(WB2, wB2, 64, 9);
  k_prep_w<<<(6912 + 127) / 128, 128, 0, stream>>>(WP,  wP,  64, 27);
  k_mask_convert<<<NVOXEL / 256, 256, 0, stream>>>(x, mask, xs, stats);

  // branch A, stage 1: conv(3,1,3) pad(1,0,1)
  k_conv<32, 3, 1, 1, 0><<<NTILE, 128, 0, stream>>>(xs, wA1, mask, hbuf, stats, nullptr);
  k_bn<<<NBN, 256, 0, stream>>>(hbuf, mask, stats, gA1, bA1, nullptr, act, nullptr, nullptr);

  // branch A, stage 2: conv(1,3,3) pad(0,1,1) -> resA (f32 chan-last)
  k_reset<<<1, 192, 0, stream>>>(stats, 0);
  k_conv<64, 1, 3, 1, 0><<<NTILE, 128, 0, stream>>>(act, wA2, mask, hbuf, stats, nullptr);
  k_bn<<<NBN, 256, 0, stream>>>(hbuf, mask, stats, gA2, bA2, nullptr, nullptr, resA, nullptr);

  // branch B, stage 1: conv(1,3,3) pad(0,1,1)
  k_reset<<<1, 192, 0, stream>>>(stats, 0);
  k_conv<32, 1, 3, 1, 0><<<NTILE, 128, 0, stream>>>(xs, wB1, mask, hbuf, stats, nullptr);
  k_bn<<<NBN, 256, 0, stream>>>(hbuf, mask, stats, gB1, bB1, nullptr, act, nullptr, nullptr);

  // branch B, stage 2: conv(3,1,3) pad(1,0,1); +resA; emit res_B (c-major) + bf16
  k_reset<<<1, 192, 0, stream>>>(stats, 0);
  k_conv<64, 3, 1, 1, 0><<<NTILE, 128, 0, stream>>>(act, wB2, mask, hbuf, stats, nullptr);
  k_bn<<<NBN, 256, 0, stream>>>(hbuf, mask, stats, gB2, bB2, resA, act, nullptr, resB);

  // pooled downsample: conv(3,3,3) pad(1,1,1) stride(2,2,1) * out_mask
  k_conv<64, 3, 3, 2, 1><<<NTILEP, 128, 0, stream>>>(act, wP, mask, nullptr, nullptr, down);
}
